// ligand_GCN_55757265436927
// MI455X (gfx1250) — matrix operand order repaired and checked by reading, not compiled, not run on hardware
//
#include <hip/hip_runtime.h>

// ---------------------------------------------------------------------------
// Problem constants (from the reference)
// ---------------------------------------------------------------------------
constexpr int NN   = 50000;    // nodes  (== 3125 * 16, exact WMMA M tiles)
constexpr int NE   = 800000;   // edges
constexpr int NG   = 1024;     // graphs
constexpr int DIN  = 78;       // input feature dim
constexpr int DINP = 96;       // DIN padded to multiple of 32 for WMMA K
constexpr int H    = 256;      // hidden 1
constexpr int H2   = 512;      // hidden 2/3
constexpr int DOUT = 128;      // output dim

typedef __bf16 bf16;
typedef bf16  v16bf __attribute__((ext_vector_type(16)));
typedef float v8f   __attribute__((ext_vector_type(8)));

// ---------------------------------------------------------------------------
// WMMA fragment loaders (CDNA5 wave32 layouts, v_wmma_f32_16x16x32_bf16).
// A 16x32 bf16: lane<16 -> M=lane, K={k0..k0+7, k0+16..k0+23}
//               lane>=16 -> M=lane-16, K={k0+8..k0+15, k0+24..k0+31}
// B 32x16 bf16 (from W^T [N,K]): lane<16 -> N=lane, K={k0..k0+7, k0+16..k0+23}
//               lane>=16 -> N=lane-16, K={k0+8..k0+15, k0+24..k0+31}
// ---------------------------------------------------------------------------
__device__ __forceinline__ v16bf load_fragA_at(const bf16* __restrict__ prow,
                                               int k0, int lane) {
  const int kb = (lane >> 4) << 3;                       // 0 or 8
  const bf16* p = prow + k0 + kb;
  union { v16bf v; uint4 q[2]; } u;
  u.q[0] = *(const uint4*)(p);                           // K offsets +0..+7
  u.q[1] = *(const uint4*)(p + 16);                      // K offsets +16..+23
  return u.v;
}

// B fragment from the LDS-staged weight strip (row-padded, bank-conflict-free)
__device__ __forceinline__ v16bf load_fragB_lds(const bf16* s, int row, int ldk,
                                                int k0, int lane) {
  const int kb = (lane >> 4) << 3;                       // 0 or 8
  const bf16* p = s + row * ldk + k0 + kb;
  union { v16bf v; uint4 q[2]; } u;
  u.q[0] = *(const uint4*)(p);                           // K +0..+7
  u.q[1] = *(const uint4*)(p + 16);                      // K +16..+23
  return u.v;
}

// ---------------------------------------------------------------------------
// Fused GraphConv GEMM: out = act( A1 @ W1t^T + [A2 @ W2t^T] + bias )
// Block = 8 waves along M sharing one 64-column weight strip, staged once in
// LDS (cooperative fill + barrier). Each wave computes 16x64 (4 accumulators);
// per k-step: global bf16 A fragment(s) + LDS B fragments -> 8 WMMAs.
// All A operands are pre-converted bf16, so the hot loop has no VALU chains.
// ---------------------------------------------------------------------------
template<int K, bool DUAL, bool RELU, typename TOUT>
__global__ __launch_bounds__(256, 2) void k_gconv_gemm(
    const bf16* __restrict__ A1, const bf16* __restrict__ W1t,
    const bf16* __restrict__ A2, const bf16* __restrict__ W2t,
    const float* __restrict__ bias, TOUT* __restrict__ out,
    int M, int N)
{
  constexpr int LDK = K + 8;                             // pad: conflict-free
  extern __shared__ bf16 smem[];
  bf16* sB1 = smem;                                      // [64][LDK]
  bf16* sB2 = smem + 64 * LDK;                           // [64][LDK] (DUAL)

  const int tid  = threadIdx.x;
  const int lane = tid & 31;
  const int wave = tid >> 5;
  const int n0   = blockIdx.y << 6;                      // 64-col strip

  // ---- cooperative LDS fill of the weight strip (all waves) ----
  constexpr int CH = K / 8;                              // 16B chunks per row
#pragma unroll 2
  for (int ci = tid; ci < 64 * CH; ci += 256) {
    const int row = ci / CH;
    const int kc  = (ci % CH) * 8;
    *(uint4*)(&sB1[row * LDK + kc]) =
        *(const uint4*)(W1t + (size_t)(n0 + row) * K + kc);
    if constexpr (DUAL)
      *(uint4*)(&sB2[row * LDK + kc]) =
          *(const uint4*)(W2t + (size_t)(n0 + row) * K + kc);
  }
  __syncthreads();

  const int m0 = (blockIdx.x * 8 + wave) << 4;           // wave-uniform
  if (m0 >= M) return;                                   // EXEC all-1s if alive
  const int nl   = lane & 15;
  const int rowA = m0 + nl;

  const bf16* pa1 = A1 + (size_t)rowA * K;
  const bf16* pa2 = A2 + (size_t)rowA * K;

  v8f c[4] = {{}, {}, {}, {}};
#pragma unroll
  for (int k0 = 0; k0 < K; k0 += 32) {
    // ---- gather: global A fragment(s) + LDS B fragments ----
    const v16bf a1 = load_fragA_at(pa1, k0, lane);
    v16bf b1f[4];
#pragma unroll
    for (int t = 0; t < 4; ++t)
      b1f[t] = load_fragB_lds(sB1, t * 16 + nl, LDK, k0, lane);
    v16bf a2;
    v16bf b2f[4];
    if constexpr (DUAL) {
      a2 = load_fragA_at(pa2, k0, lane);
#pragma unroll
      for (int t = 0; t < 4; ++t)
        b2f[t] = load_fragB_lds(sB2, t * 16 + nl, LDK, k0, lane);
    }
    // ---- compute: 8 back-to-back WMMAs ----
#pragma unroll
    for (int t = 0; t < 4; ++t)
      c[t] = __builtin_amdgcn_wmma_f32_16x16x32_bf16(false, a1, false, b1f[t],
                                                     (short)0, c[t], false, false);
    if constexpr (DUAL) {
#pragma unroll
      for (int t = 0; t < 4; ++t)
        c[t] = __builtin_amdgcn_wmma_f32_16x16x32_bf16(false, a2, false, b2f[t],
                                                       (short)0, c[t], false, false);
    }
  }

  // D layout: VGPR r -> M = m0 + r (+8 for lanes 16..31), N = n0 + t*16 + nl
  const int rbase = m0 + ((lane >> 4) << 3);
#pragma unroll
  for (int t = 0; t < 4; ++t) {
    const int col = n0 + t * 16 + nl;
    const float bv = bias[col];
#pragma unroll
    for (int r = 0; r < 8; ++r) {
      float v = c[t][r] + bv;
      if constexpr (RELU) v = fmaxf(v, 0.0f);
      out[(size_t)(rbase + r) * N + col] = (TOUT)v;
    }
  }
}

// ---------------------------------------------------------------------------
// Helpers: zero, streaming f32->bf16 convert, input pad, weight prep
// ---------------------------------------------------------------------------
__global__ void k_zero(float* __restrict__ p, long long n) {
  long long i = (long long)blockIdx.x * blockDim.x + threadIdx.x;
  if (i < n) p[i] = 0.0f;
}

// n4 = n/4 groups; float4 in -> 4 bf16 (8B) out
__global__ void k_cvtbf16(const float* __restrict__ in, bf16* __restrict__ out,
                          long long n4) {
  long long g = (long long)blockIdx.x * blockDim.x + threadIdx.x;
  if (g >= n4) return;
  const float4 v = ((const float4*)in)[g];
  union { bf16 b[4]; uint2 u; } o;
  o.b[0] = (bf16)v.x; o.b[1] = (bf16)v.y; o.b[2] = (bf16)v.z; o.b[3] = (bf16)v.w;
  ((uint2*)out)[g] = o.u;
}

__global__ void k_padx(const float* __restrict__ x, bf16* __restrict__ xp) {
  long long gid = (long long)blockIdx.x * blockDim.x + threadIdx.x;
  if (gid >= (long long)NN * DINP) return;
  const int r = (int)(gid / DINP), c = (int)(gid % DINP);
  xp[gid] = (c < DIN) ? (bf16)x[(size_t)r * DIN + c] : (bf16)0.0f;
}

// W [K,N] f32 row-major  ->  Wt [N,Kp] bf16 (zero-padded K)
__global__ void k_wprep(const float* __restrict__ W, bf16* __restrict__ Wt,
                        int K, int N, int Kp) {
  long long gid = (long long)blockIdx.x * blockDim.x + threadIdx.x;
  if (gid >= (long long)N * Kp) return;
  const int n = (int)(gid / Kp), k = (int)(gid % Kp);
  Wt[gid] = (k < K) ? (bf16)W[(size_t)k * N + n] : (bf16)0.0f;
}

// ---------------------------------------------------------------------------
// Edge scatter-add (segment_sum over edges): aggr[dst] += feat[src]
// A wave32 covers one edge's contiguous feature chunk (uniform src/dst load,
// 512B coalesced gather); fp32 atomics resolve mostly in the 192MB L2.
// ---------------------------------------------------------------------------
__global__ __launch_bounds__(256) void k_scatter_f32_pad(
    const float* __restrict__ x, const int* __restrict__ src,
    const int* __restrict__ dst, float* __restrict__ aggr)
{
  long long gid = (long long)blockIdx.x * blockDim.x + threadIdx.x;
  if (gid >= (long long)NE * 20) return;                  // 20 groups of 4 cols
  const int e = (int)(gid / 20), g = (int)(gid % 20);
  const int s = src[e], d = dst[e];
  const int c0 = g * 4;
#pragma unroll
  for (int j = 0; j < 4; ++j) {
    const int c = c0 + j;
    if (c < DIN)
      atomicAdd(&aggr[(size_t)d * DINP + c], x[(size_t)s * DIN + c]);
  }
}

__global__ __launch_bounds__(256) void k_scatter_bf16(
    const bf16* __restrict__ h, const int* __restrict__ src,
    const int* __restrict__ dst, float* __restrict__ aggr, int dim)
{
  const int groups = dim >> 3;                            // 8 bf16 per thread
  long long gid = (long long)blockIdx.x * blockDim.x + threadIdx.x;
  if (gid >= (long long)NE * groups) return;
  const int e = (int)(gid / groups), g = (int)(gid % groups);
  const int s = src[e], d = dst[e];
  const uint4 raw = *(const uint4*)(h + (size_t)s * dim + g * 8);
  const bf16* b = (const bf16*)&raw;
  float* q = aggr + (size_t)d * dim + g * 8;
#pragma unroll
  for (int j = 0; j < 8; ++j) atomicAdd(q + j, (float)b[j]);
}

// ---------------------------------------------------------------------------
// LayerNorm(128) + ReLU + graph pooling. One wave32 per row (4 floats/lane).
// ---------------------------------------------------------------------------
__global__ __launch_bounds__(256) void k_ln_pool(
    const float* __restrict__ z, const float* __restrict__ gamma,
    const float* __restrict__ beta, const int* __restrict__ batch,
    float* __restrict__ xatom, float* __restrict__ pooled)
{
  const int lane = threadIdx.x & 31;
  const int row  = blockIdx.x * 8 + (threadIdx.x >> 5);
  if (row >= NN) return;
  const float4 v = *(const float4*)(z + (size_t)row * DOUT + lane * 4);
  float s  = v.x + v.y + v.z + v.w;
  float ss = v.x * v.x + v.y * v.y + v.z * v.z + v.w * v.w;
#pragma unroll
  for (int off = 16; off > 0; off >>= 1) {
    s  += __shfl_xor(s,  off, 32);
    ss += __shfl_xor(ss, off, 32);
  }
  const float mu   = s * (1.0f / DOUT);
  const float var  = ss * (1.0f / DOUT) - mu * mu;
  const float rstd = rsqrtf(var + 1e-5f);
  const int c  = lane * 4;
  const int bg = batch[row];
  const float in[4] = {v.x, v.y, v.z, v.w};
  float4 r;
  float* rp = &r.x;
#pragma unroll
  for (int j = 0; j < 4; ++j) {
    float t = (in[j] - mu) * rstd * gamma[c + j] + beta[c + j];
    t = fmaxf(t, 0.0f);
    rp[j] = t;
    atomicAdd(&pooled[(size_t)bg * DOUT + c + j], t);
  }
  *(float4*)(xatom + (size_t)row * DOUT + c) = r;
}

// ---------------------------------------------------------------------------
// Host-side launch
// ---------------------------------------------------------------------------
static inline unsigned nblk(long long total, int b) {
  return (unsigned)((total + b - 1) / b);
}

extern "C" void kernel_launch(void* const* d_in, const int* in_sizes, int n_in,
                              void* d_out, int out_size, void* d_ws, size_t ws_size,
                              hipStream_t stream) {
  (void)in_sizes; (void)n_in; (void)out_size; (void)ws_size;

  // ---- inputs (setup_inputs dict order) ----
  const float* x      = (const float*)d_in[0];
  const int*   ei     = (const int*)  d_in[1];   // [2, NE]
  const int*   batch  = (const int*)  d_in[2];
  const float* W1rel  = (const float*)d_in[3];
  const float* b1     = (const float*)d_in[4];
  const float* W1root = (const float*)d_in[5];
  const float* W2rel  = (const float*)d_in[6];
  const float* b2     = (const float*)d_in[7];
  const float* W2root = (const float*)d_in[8];
  const float* W3rel  = (const float*)d_in[9];
  const float* b3     = (const float*)d_in[10];
  const float* W3root = (const float*)d_in[11];
  const float* Wfc    = (const float*)d_in[12];
  const float* bfc    = (const float*)d_in[13];
  const float* lng    = (const float*)d_in[14];
  const float* lnb    = (const float*)d_in[15];

  const int* src = ei;
  const int* dst = ei + NE;

  float* xatom  = (float*)d_out;                          // [NN, DOUT]
  float* pooled = (float*)d_out + (size_t)NN * DOUT;      // [NG, DOUT]

  // ---- workspace carve-out (256B aligned slices) ----
  char*  w   = (char*)d_ws;
  size_t off = 0;
  auto carve = [&](size_t bytes) {
    void* p = w + off;
    off = (off + bytes + 255) & ~(size_t)255;
    return p;
  };
  bf16*  xpad    = (bf16*) carve((size_t)NN * DINP * 2);
  float* aggr1   = (float*)carve((size_t)NN * DINP * 4);
  bf16*  aggr1b  = (bf16*) carve((size_t)NN * DINP * 2);
  bf16*  h1      = (bf16*) carve((size_t)NN * H    * 2);
  float* aggr2   = (float*)carve((size_t)NN * H    * 4);
  bf16*  aggr2b  = (bf16*) carve((size_t)NN * H    * 2);
  bf16*  h2      = (bf16*) carve((size_t)NN * H2   * 2);
  float* aggr3   = (float*)carve((size_t)NN * H2   * 4);
  bf16*  aggr3b  = (bf16*) carve((size_t)NN * H2   * 2);
  bf16*  h3      = (bf16*) carve((size_t)NN * H2   * 2);
  float* z       = (float*)carve((size_t)NN * DOUT * 4);
  bf16*  w1relt  = (bf16*) carve((size_t)H  * DINP * 2);
  bf16*  w1roott = (bf16*) carve((size_t)H  * DINP * 2);
  bf16*  w2relt  = (bf16*) carve((size_t)H2 * H    * 2);
  bf16*  w2roott = (bf16*) carve((size_t)H2 * H    * 2);
  bf16*  w3relt  = (bf16*) carve((size_t)H2 * H2   * 2);
  bf16*  w3roott = (bf16*) carve((size_t)H2 * H2   * 2);
  bf16*  wfct    = (bf16*) carve((size_t)DOUT * H2 * 2);

  const dim3 blk(256);
  const unsigned gx = (unsigned)((NN / 16 + 7) / 8);      // 391 blocks along M

  // dynamic LDS per GEMM: strips of 64 x (K+8) bf16 (x2 when DUAL)
  const size_t sh1  = (size_t)2 * 64 * (DINP + 8) * sizeof(bf16);
  const size_t sh2  = (size_t)2 * 64 * (H    + 8) * sizeof(bf16);
  const size_t sh3  = (size_t)2 * 64 * (H2   + 8) * sizeof(bf16);
  const size_t shfc = (size_t)1 * 64 * (H2   + 8) * sizeof(bf16);

  // ---- zero the atomic accumulation targets ----
  k_zero<<<nblk((long long)NN * DINP, 256), blk, 0, stream>>>(aggr1, (long long)NN * DINP);
  k_zero<<<nblk((long long)NN * H,    256), blk, 0, stream>>>(aggr2, (long long)NN * H);
  k_zero<<<nblk((long long)NN * H2,   256), blk, 0, stream>>>(aggr3, (long long)NN * H2);
  k_zero<<<nblk((long long)NG * DOUT, 256), blk, 0, stream>>>(pooled, (long long)NG * DOUT);

  // ---- one-time per-call prep: pad x (bf16), transpose+bf16 weights ----
  k_padx<<<nblk((long long)NN * DINP, 256), blk, 0, stream>>>(x, xpad);
  k_wprep<<<nblk((long long)H  * DINP, 256), blk, 0, stream>>>(W1rel,  w1relt,  DIN, H,  DINP);
  k_wprep<<<nblk((long long)H  * DINP, 256), blk, 0, stream>>>(W1root, w1roott, DIN, H,  DINP);
  k_wprep<<<nblk((long long)H2 * H,    256), blk, 0, stream>>>(W2rel,  w2relt,  H,   H2, H);
  k_wprep<<<nblk((long long)H2 * H,    256), blk, 0, stream>>>(W2root, w2roott, H,   H2, H);
  k_wprep<<<nblk((long long)H2 * H2,   256), blk, 0, stream>>>(W3rel,  w3relt,  H2,  H2, H2);
  k_wprep<<<nblk((long long)H2 * H2,   256), blk, 0, stream>>>(W3root, w3roott, H2,  H2, H2);
  k_wprep<<<nblk((long long)DOUT * H2, 256), blk, 0, stream>>>(Wfc,    wfct,    H2, DOUT, H2);

  // ---- layer 1: scatter (fp32) -> bf16 -> fused dual-GEMM+ReLU -> h1 ----
  k_scatter_f32_pad<<<nblk((long long)NE * 20, 256), blk, 0, stream>>>(x, src, dst, aggr1);
  k_cvtbf16<<<nblk((long long)NN * DINP / 4, 256), blk, 0, stream>>>(aggr1, aggr1b,
                                                                     (long long)NN * DINP / 4);
  k_gconv_gemm<DINP, true, true, bf16>
      <<<dim3(gx, H / 64), blk, sh1, stream>>>(aggr1b, w1relt, xpad, w1roott, b1, h1, NN, H);

  // ---- layer 2 ----
  k_scatter_bf16<<<nblk((long long)NE * (H / 8), 256), blk, 0, stream>>>(h1, src, dst, aggr2, H);
  k_cvtbf16<<<nblk((long long)NN * H / 4, 256), blk, 0, stream>>>(aggr2, aggr2b,
                                                                  (long long)NN * H / 4);
  k_gconv_gemm<H, true, true, bf16>
      <<<dim3(gx, H2 / 64), blk, sh2, stream>>>(aggr2b, w2relt, h1, w2roott, b2, h2, NN, H2);

  // ---- layer 3 ----
  k_scatter_bf16<<<nblk((long long)NE * (H2 / 8), 256), blk, 0, stream>>>(h2, src, dst, aggr3, H2);
  k_cvtbf16<<<nblk((long long)NN * H2 / 4, 256), blk, 0, stream>>>(aggr3, aggr3b,
                                                                   (long long)NN * H2 / 4);
  k_gconv_gemm<H2, true, true, bf16>
      <<<dim3(gx, H2 / 64), blk, sh3, stream>>>(aggr3b, w3relt, h2, w3roott, b3, h3, NN, H2);

  // ---- fc (no relu before LN) -> z (fp32) ----
  k_gconv_gemm<H2, false, false, float>
      <<<dim3(gx, DOUT / 64), blk, shfc, stream>>>(h3, wfct, h3, wfct, bfc, z, NN, DOUT);

  // ---- LayerNorm + ReLU + pool ----
  k_ln_pool<<<NN / 8, blk, 0, stream>>>(z, lng, lnb, batch, xatom, pooled);
}
